// DistortionAttention_38603166057144
// MI455X (gfx1250) — matrix operand order repaired and checked by reading, not compiled, not run on hardware
//
#include <hip/hip_runtime.h>

// ---------------------------------------------------------------------------
// Types for CDNA5 WMMA (wave32, 16x16x32 bf16 -> f32)
// ---------------------------------------------------------------------------
typedef __attribute__((ext_vector_type(16))) __bf16 v16bf;
typedef __attribute__((ext_vector_type(8)))  float  v8f;
typedef __attribute__((ext_vector_type(4)))  unsigned int v4u;
typedef __attribute__((ext_vector_type(8)))  int v8i;
typedef __attribute__((ext_vector_type(4)))  int v4i;

union AB  { v16bf v; unsigned int u[8]; };

__device__ __forceinline__ unsigned short f2bf(float f) {
  unsigned int x = __float_as_uint(f);
  unsigned int r = x + 0x7FFFu + ((x >> 16) & 1u);   // round-to-nearest-even
  return (unsigned short)(r >> 16);
}
__device__ __forceinline__ unsigned short f2bf_trunc(float f) {  // cheap truncation (hot path)
  return (unsigned short)(__float_as_uint(f) >> 16);
}
__device__ __forceinline__ unsigned int pack2(float lo, float hi) {
  return (unsigned int)f2bf(lo) | ((unsigned int)f2bf(hi) << 16);
}
__device__ __forceinline__ v8f wmma_bf16(const AB& a, const AB& b, v8f c) {
  return __builtin_amdgcn_wmma_f32_16x16x32_bf16(
      /*neg_a=*/false, a.v, /*neg_b=*/false, b.v,
      /*c_mod=*/(short)0, c, /*reuse_a=*/false, /*reuse_b=*/false);
}

// wave32 xor-lane shuffle via ds_swizzle (immediate pattern, no address VGPRs)
#define SWIZ_XOR(v, mask) \
  __uint_as_float((unsigned)__builtin_amdgcn_ds_swizzle((int)__float_as_uint(v), 0x1F | ((mask) << 10)))

#define BN   4
#define CC   256
#define HW   4096
#define QKC  32
#define ND   58
#define ATT_WAVES 4

#define HAVE_TDM __has_builtin(__builtin_amdgcn_tensor_load_to_lds)

__device__ __forceinline__ void wait_tensorcnt0() {
#if __has_builtin(__builtin_amdgcn_s_wait_tensorcnt)
  __builtin_amdgcn_s_wait_tensorcnt(0);
#else
  asm volatile("s_wait_tensorcnt 0x0" ::: "memory");
#endif
}

// ---------------------------------------------------------------------------
// K0: zero helper
// ---------------------------------------------------------------------------
__global__ void k_zero(float* p, int n) {
  int i = blockIdx.x * 256 + threadIdx.x;
  if (i < n) p[i] = 0.f;
}

// ---------------------------------------------------------------------------
// Weight prep: fp32 [OC][IC] -> pair-packed bf16 dwords [OC][IC/2]
// ---------------------------------------------------------------------------
__global__ void k_pack_w(const float* __restrict__ W, unsigned int* __restrict__ out, int n_pairs) {
  int i = blockIdx.x * 256 + threadIdx.x;
  if (i < n_pairs) out[i] = pack2(W[2*i], W[2*i+1]);
}

// Weight prep 3x3: fp32 [OC][IC][3][3] -> [tap][OC][IC/2] pair-packed dwords
__global__ void k_pack_w3(const float* __restrict__ W, unsigned int* __restrict__ out,
                          int OC, int IC) {
  int i = blockIdx.x * 256 + threadIdx.x;            // over 9*OC*(IC/2)
  int icp = IC >> 1;
  int n = 9 * OC * icp;
  if (i >= n) return;
  int t   = i / (OC * icp);
  int rem = i - t * OC * icp;
  int oc  = rem / icp;
  int ip  = rem - oc * icp;
  out[i] = pack2(W[(oc * IC + 2*ip    ) * 9 + t],
                 W[(oc * IC + 2*ip + 1) * 9 + t]);
}

// ---------------------------------------------------------------------------
// K1: xa = conv1x1(x, w_adj) + b_adj ; fp32 [b][c][hw] + packed xpk [b][c/2][hw]
// ---------------------------------------------------------------------------
__global__ void k_adjust(const float* __restrict__ x, const float* __restrict__ w,
                         const float* __restrict__ bias,
                         float* __restrict__ xa, unsigned int* __restrict__ xpk) {
  int idx = blockIdx.x * 256 + threadIdx.x;          // over B*128*HW = 2M
  int px = idx & (HW - 1);
  int cp = (idx >> 12) & 127;
  int b  = idx >> 19;
  int c0 = 2 * cp, c1 = c0 + 1;
  const float* xb = x + b * 3 * HW;
  float i0 = xb[px], i1 = xb[HW + px], i2 = xb[2*HW + px];
  float v0 = bias[c0] + w[c0*3+0]*i0 + w[c0*3+1]*i1 + w[c0*3+2]*i2;
  float v1 = bias[c1] + w[c1*3+0]*i0 + w[c1*3+1]*i1 + w[c1*3+2]*i2;
  xa[((size_t)b * CC + c0) * HW + px] = v0;
  xa[((size_t)b * CC + c1) * HW + px] = v1;
  xpk[idx] = pack2(v0, v1);
}

// ---------------------------------------------------------------------------
// K2: 1x1-conv GEMM via WMMA, register-blocked over NT oc-tiles.
//     MODE 0: v -> bf16 [b][oc][px];  MODE 1: q -> packed [b][px][oc/2];
//     MODE 2: k -> packed [b][oc/2][px]
// ---------------------------------------------------------------------------
template<int NT, int MODE>
__global__ void __launch_bounds__(32, 1)
k_gemm(const unsigned int* __restrict__ Wpk, const float* __restrict__ bias,
       const unsigned int* __restrict__ xpk, void* __restrict__ outp, int OC) {
  int lane = threadIdx.x;
  int half = lane >> 4;
  int l15  = lane & 15;
  int pt = blockIdx.x, otb = blockIdx.y * NT, b = blockIdx.z;
  int col = pt * 16 + l15;
  const unsigned int* xb = xpk + (size_t)b * (CC/2) * HW;
  v8f acc[NT];
  v8f zero = {};
  #pragma unroll
  for (int t = 0; t < NT; ++t) acc[t] = zero;
  for (int kc = 0; kc < CC; kc += 32) {
    AB bt;
    #pragma unroll
    for (int j = 0; j < 8; ++j)                      // B loaded once per k-chunk
      bt.u[j] = xb[((kc >> 1) + (half ? 8 : 0) + j) * HW + col];
    #pragma unroll
    for (int t = 0; t < NT; ++t) {
      AB a;
      int rowm = (otb + t) * 16 + l15;
      const uint4* wr4 = (const uint4*)(Wpk + rowm * (CC/2) + (kc >> 1));
      *(uint4*)&a.u[0] = wr4[half ? 2 : 0];
      *(uint4*)&a.u[4] = wr4[half ? 3 : 1];
      acc[t] = wmma_bf16(a, bt, acc[t]);
    }
  }
  #pragma unroll
  for (int t = 0; t < NT; ++t) {
    int ot = otb + t;
    if (MODE == 0) {
      unsigned short* o = (unsigned short*)outp;
      #pragma unroll
      for (int r = 0; r < 8; ++r) {
        int oc = ot * 16 + r + half * 8;
        o[((size_t)b * OC + oc) * HW + col] = f2bf(acc[t][r] + bias[oc]);
      }
    } else if (MODE == 1) {
      unsigned int* o = (unsigned int*)outp;
      #pragma unroll
      for (int i = 0; i < 4; ++i) {
        int oc = ot * 16 + half * 8 + 2 * i;
        o[((size_t)b * HW + col) * (OC/2) + ot * 8 + half * 4 + i] =
            pack2(acc[t][2*i] + bias[oc], acc[t][2*i+1] + bias[oc+1]);
      }
    } else {
      unsigned int* o = (unsigned int*)outp;
      #pragma unroll
      for (int i = 0; i < 4; ++i) {
        int oc = ot * 16 + half * 8 + 2 * i;
        o[((size_t)b * (OC/2) + ot * 8 + half * 4 + i) * HW + col] =
            pack2(acc[t][2*i] + bias[oc], acc[t][2*i+1] + bias[oc+1]);
      }
    }
  }
}

// ---------------------------------------------------------------------------
// K3: 3x3 conv implicit GEMM via WMMA, register-blocked over NT oc-tiles
// ---------------------------------------------------------------------------
template<int NT>
__global__ void __launch_bounds__(32, 1)
k_conv3x3(const unsigned int* __restrict__ W3pk, const float* __restrict__ bias,
          const unsigned int* __restrict__ inpk,
          unsigned int* __restrict__ outpk, float* __restrict__ pooled,
          int IC, int OC, int do_pool) {
  int lane = threadIdx.x;
  int half = lane >> 4;
  int l15  = lane & 15;
  int pt = blockIdx.x, otb = blockIdx.y * NT, b = blockIdx.z;
  int pix  = pt * 16 + l15;
  int py = pix >> 6, pxx = pix & 63;
  const unsigned int* ib = inpk + (size_t)b * (IC/2) * HW;
  v8f acc[NT];
  v8f zero = {};
  #pragma unroll
  for (int t = 0; t < NT; ++t) acc[t] = zero;
  for (int t = 0; t < 9; ++t) {
    int dy = t / 3 - 1, dx = t % 3 - 1;
    int ny = py + dy, nx = pxx + dx;
    bool valid = ((unsigned)ny < 64u) && ((unsigned)nx < 64u);
    int pix2 = valid ? (ny * 64 + nx) : 0;
    for (int kc = 0; kc < IC; kc += 32) {
      AB bt;
      #pragma unroll
      for (int j = 0; j < 8; ++j) {
        unsigned int u = ib[((kc >> 1) + (half ? 8 : 0) + j) * HW + pix2];
        bt.u[j] = valid ? u : 0u;                    // zero-pad border, value-level
      }
      #pragma unroll
      for (int tt = 0; tt < NT; ++tt) {
        AB a;
        int rowm = (otb + tt) * 16 + l15;
        const uint4* wr4 = (const uint4*)(W3pk + (size_t)(t * OC + rowm) * (IC/2) + (kc >> 1));
        *(uint4*)&a.u[0] = wr4[half ? 2 : 0];
        *(uint4*)&a.u[4] = wr4[half ? 3 : 1];
        acc[tt] = wmma_bf16(a, bt, acc[tt]);
      }
    }
  }
  #pragma unroll
  for (int tt = 0; tt < NT; ++tt) {
    int ot = otb + tt;
    float v[8];
    #pragma unroll
    for (int r = 0; r < 8; ++r) {
      int oc = ot * 16 + r + half * 8;
      float f = acc[tt][r] + bias[oc];
      v[r] = f > 0.f ? f : 0.f;                      // ReLU
    }
    if (outpk) {
      #pragma unroll
      for (int i = 0; i < 4; ++i)
        outpk[((size_t)b * (OC/2) + ot * 8 + half * 4 + i) * HW + pix] = pack2(v[2*i], v[2*i+1]);
    }
    if (do_pool) {
      #pragma unroll
      for (int r = 0; r < 8; ++r) {
        float s = v[r];
        s += SWIZ_XOR(s, 1);
        s += SWIZ_XOR(s, 2);
        s += SWIZ_XOR(s, 4);
        s += SWIZ_XOR(s, 8);
        if (l15 == 0) atomicAdd(&pooled[b * OC + ot * 16 + r + half * 8], s);
      }
    }
  }
}

// ---------------------------------------------------------------------------
// K4: logits + argmax routing
// ---------------------------------------------------------------------------
__global__ void k_logits(const float* __restrict__ pooled, const float* __restrict__ wfc,
                         const float* __restrict__ bfc, float* __restrict__ out_logits,
                         int* __restrict__ dt) {
  __shared__ float lg[BN][ND];
  int tid = threadIdx.x;
  if (tid < BN * ND) {
    int b = tid / ND, d = tid % ND;
    float s = bfc[d];
    for (int k = 0; k < 128; ++k)
      s += (pooled[b * 128 + k] * (1.f / 4096.f)) * wfc[d * 128 + k];
    lg[b][d] = s;
    out_logits[b * ND + d] = s;
  }
  __syncthreads();
  if (tid < BN) {
    int best = 0; float bv = lg[tid][0];
    for (int d = 1; d < ND; ++d) { float v = lg[tid][d]; if (v > bv) { bv = v; best = d; } }
    dt[tid] = best;
  }
}

// ---------------------------------------------------------------------------
// K5: fused experts: sobel-magnitude mean + HSV saturation (one channel pass)
// ---------------------------------------------------------------------------
__global__ void k_experts(const float* __restrict__ xa, float* __restrict__ sobel_out,
                          float* __restrict__ sat_out) {
  int idx = blockIdx.x * 256 + threadIdx.x;          // over B*HW
  int b = idx >> 12, pix = idx & (HW - 1);
  int py = pix >> 6, px = pix & 63;
  const float* xb = xa + (size_t)b * CC * HW;
  float msum = 0.f, mx = -1e30f, mn = 1e30f;
  for (int c = 0; c < CC; ++c) {
    const float* p = xb + (size_t)c * HW;
    float v[3][3];
    #pragma unroll
    for (int dy = -1; dy <= 1; ++dy)
      #pragma unroll
      for (int dx = -1; dx <= 1; ++dx) {
        int ny = py + dy, nx = px + dx;
        v[dy+1][dx+1] = (((unsigned)ny < 64u) && ((unsigned)nx < 64u)) ? p[ny*64+nx] : 0.f;
      }
    float gx = -v[0][0] + v[0][2] - 2.f*v[1][0] + 2.f*v[1][2] - v[2][0] + v[2][2];
    float gy = -v[0][0] - 2.f*v[0][1] - v[0][2] + v[2][0] + 2.f*v[2][1] + v[2][2];
    msum += sqrtf(gx*gx + gy*gy);
    float cv = v[1][1];
    mx = fmaxf(mx, cv); mn = fminf(mn, cv);
  }
  float m = msum * (1.f / 256.f);
  sobel_out[idx] = 1.f / (1.f + __expf(-m));
  sat_out[idx]   = (mx - mn) / fmaxf(mx, 1e-6f);
}

// ---------------------------------------------------------------------------
// K6: 64x64 2D DFT magnitude, accumulated over channels (twiddle-table in LDS)
// ---------------------------------------------------------------------------
__global__ void k_fft(const float* __restrict__ xa, float* __restrict__ fmag_sum) {
  __shared__ float sinp[64][65];
  __shared__ float sre[64][65];
  __shared__ float sim[64][65];
  __shared__ float twr[64], twi[64];
  int t = threadIdx.x;                               // 64 threads
  int bc = blockIdx.x;
  int b = bc >> 8, c = bc & 255;
  const float* p = xa + ((size_t)b * CC + c) * HW;
  if (t < 64) {
    float ang = -6.283185307179586f * (float)t / 64.f;
    twr[t] = __cosf(ang); twi[t] = __sinf(ang);
  }
  for (int i = t; i < HW; i += 64) sinp[i >> 6][i & 63] = p[i];
  __syncthreads();
  for (int kx = 0; kx < 64; ++kx) {                  // row DFT, thread = row y
    float ar = 0.f, ai = 0.f;
    for (int x = 0; x < 64; ++x) {
      int n = (kx * x) & 63;
      float xv = sinp[t][x];
      ar += xv * twr[n];
      ai += xv * twi[n];
    }
    sre[t][kx] = ar; sim[t][kx] = ai;
  }
  __syncthreads();
  for (int ky = 0; ky < 64; ++ky) {                  // col DFT, thread = column
    float ar = 0.f, ai = 0.f;
    for (int y = 0; y < 64; ++y) {
      int n = (ky * y) & 63;
      float rr = sre[y][t], ii = sim[y][t];
      ar += rr * twr[n] - ii * twi[n];
      ai += rr * twi[n] + ii * twr[n];
    }
    atomicAdd(&fmag_sum[b * HW + ky * 64 + t], sqrtf(ar*ar + ai*ai));
  }
}

// ---------------------------------------------------------------------------
// K7: per-sample routed dispatch -> d_out
// ---------------------------------------------------------------------------
__global__ void k_filtered(const float* __restrict__ xa, const float* __restrict__ sobel_out,
                           const float* __restrict__ sat_out, const float* __restrict__ fmag_sum,
                           const int* __restrict__ dt, float* __restrict__ out) {
  int idx = blockIdx.x * 256 + threadIdx.x;          // 4M
  int pix = idx & (HW - 1);
  int b = idx >> 20;
  int d = dt[b];
  unsigned lo = (d < 32) ? (1u << d) : 0u;
  unsigned hi = (d >= 32) ? (1u << (d - 32)) : 0u;
  const unsigned SOB_LO = 0x000FFFFFu;               // 0..19
  const unsigned SOB_HI = 0x02001FF8u;               // 35..44, 57
  const unsigned HSV_LO = 0xFFC00000u;               // 22..31
  const unsigned HSV_HI = 0x00000007u;               // 32..34
  const unsigned FOUR_HI= 0x01FFE000u;               // 45..56
  float v;
  if      ((lo & SOB_LO) | (hi & SOB_HI)) v = sobel_out[b * HW + pix];
  else if ((lo & HSV_LO) | (hi & HSV_HI)) v = sat_out[b * HW + pix];
  else if (hi & FOUR_HI)                  v = 1.f / (1.f + __expf(-fmag_sum[b * HW + pix] * (1.f / 256.f)));
  else                                    v = xa[idx];
  out[idx] = v;
}

// ---------------------------------------------------------------------------
// K8: flash attention. 4 waves / block share K,V m-chunks through LDS.
//     V chunk staged by the Tensor Data Mover: one 2D TDM descriptor
//     (tile 32m x 128c rows, stride 4096) lands row-packed in LDS, exactly
//     matching the vs[c][16-dword] B-operand layout; wave 0 issues, waits on
//     TENSORcnt, barrier publishes. Fallback: manual b32 staging.
// ---------------------------------------------------------------------------
__global__ void __launch_bounds__(32 * ATT_WAVES, 1)
k_attn(const unsigned int* __restrict__ qpk,
       const unsigned int* __restrict__ kpk,
       const unsigned short* __restrict__ vb,
       float* __restrict__ out) {
  __shared__ unsigned int   vs[128 * 16];            // 8 KB: this block's 128 channels
  __shared__ unsigned int   ks[32 * 16];             // 2 KB
  __shared__ unsigned short pl[ATT_WAVES][16 * 32];  // 4 KB
  int tid  = threadIdx.x;
  int wv   = tid >> 5;
  int lane = tid & 31;
  int half = lane >> 4;
  int l15  = lane & 15;
  int b   = blockIdx.z;
  int ch0 = blockIdx.y * 8;                          // c-tile base (8 tiles = 128 ch)
  int n0  = (blockIdx.x * ATT_WAVES + wv) * 16;

  AB qa;                                             // Q rows for this wave, 2x b128
  {
    const uint4* qr = (const uint4*)(qpk + ((size_t)b * HW + n0 + l15) * (QKC/2));
    *(uint4*)&qa.u[0] = qr[half ? 2 : 0];
    *(uint4*)&qa.u[4] = qr[half ? 3 : 1];
  }

  float mrun[8], lrun[8];
  v8f acc[8];
  v8f zero = {};
  #pragma unroll
  for (int r = 0; r < 8; ++r) { mrun[r] = -1e30f; lrun[r] = 0.f; }
  #pragma unroll
  for (int ct = 0; ct < 8; ++ct) acc[ct] = zero;

  const float scale = 0.17677669529663687f;          // 32^-0.5
  const unsigned int*   kg = kpk + (size_t)b * (QKC/2) * HW;
  const unsigned short* vrow0 = vb + (size_t)b * CC * HW + (size_t)ch0 * 16 * HW;
  const unsigned int*   vg = (const unsigned int*)vrow0;   // dword rows [c][2048]

  for (int m0 = 0; m0 < HW; m0 += 32) {
    // ---- stage V chunk (TDM DMA if available) + K chunk ----
#if HAVE_TDM
    if (wv == 0) {
      unsigned long long ga = (unsigned long long)(const void*)(vrow0 + m0);
      unsigned lds_off = (unsigned)(size_t)(void*)vs;
      v4u g0 = { 1u,                                         // count=1 valid user D#
                 lds_off,                                     // lds_addr[31:0]
                 (unsigned)(ga & 0xFFFFFFFFu),                // global_addr[31:0]
                 (unsigned)(((ga >> 32) & 0x01FFFFFFu) | (2u << 30)) }; // addr[56:32] | type=2
      v8i g1 = { (int)(1u << 16),                             // data_size = 2 bytes
                 (int)((HW & 0xFFFFu) << 16),                 // tensor_dim0 lo16 (4096)
                 (int)(((unsigned)HW >> 16) | (128u << 16)),  // dim0 hi16 | tensor_dim1 lo16 (128)
                 (int)(32u << 16),                            // dim1 hi16 | tile_dim0 = 32
                 (int)128,                                    // tile_dim1 = 128, tile_dim2 = 0
                 (int)HW,                                     // tensor_dim0_stride lo32 (4096)
                 0, 0 };                                      // stride hi | dim1_stride
      v4i gz4 = {0, 0, 0, 0};
      v8i gz8 = {0, 0, 0, 0, 0, 0, 0, 0};
      // 6-arg form (clang-23 / therock): (g0, g1, g2, g3, g4, cpol)
      __builtin_amdgcn_tensor_load_to_lds(g0, g1, gz4, gz4, gz8, 0);
      wait_tensorcnt0();                                      // DMA complete before barrier
    }
#else
    {
      int c0 = tid >> 4, mp0 = tid & 15;
      #pragma unroll
      for (int i = 0; i < (128 * 16) / 128; ++i) {   // 16 dwords per thread
        int c = c0 + i * 8;
        vs[c * 16 + mp0] = vg[c * (HW/2) + (m0 >> 1) + mp0];
      }
      if (m0 + 32 < HW)
        __builtin_prefetch(vg + c0 * (HW/2) + ((m0 + 32) >> 1) + mp0, 0, 1);
    }
#endif
    {
      #pragma unroll
      for (int i = 0; i < 512 / 128; ++i) {          // ks: [m][pair] transposed stage
        int idx = tid + i * 128;
        int m = idx >> 4, p = idx & 15;
        ks[idx] = kg[p * HW + m0 + m];
      }
      if (m0 + 32 < HW)                              // gfx1250 global_prefetch of next K chunk
        __builtin_prefetch(kg + (tid & 15) * HW + m0 + 32 + (tid >> 4), 0, 1);
    }
    __syncthreads();

    // ---- two score tiles from LDS (2x ds_load_b128 each) ----
    AB bt0, bt1;
    {
      const uint4* kr0 = (const uint4*)(ks + l15 * 16);
      const uint4* kr1 = (const uint4*)(ks + (16 + l15) * 16);
      *(uint4*)&bt0.u[0] = kr0[half ? 2 : 0];
      *(uint4*)&bt0.u[4] = kr0[half ? 3 : 1];
      *(uint4*)&bt1.u[0] = kr1[half ? 2 : 0];
      *(uint4*)&bt1.u[4] = kr1[half ? 3 : 1];
    }
    v8f s0 = wmma_bf16(qa, bt0, zero);
    v8f s1 = wmma_bf16(qa, bt1, zero);

    float alpha[8];
    #pragma unroll
    for (int r = 0; r < 8; ++r) {                    // row n = r + 8*half
      float a0 = s0[r] * scale, a1 = s1[r] * scale;
      float sm = fmaxf(a0, a1);
      sm = fmaxf(sm, SWIZ_XOR(sm, 1));
      sm = fmaxf(sm, SWIZ_XOR(sm, 2));
      sm = fmaxf(sm, SWIZ_XOR(sm, 4));
      sm = fmaxf(sm, SWIZ_XOR(sm, 8));
      float mnew = fmaxf(mrun[r], sm);
      alpha[r] = __expf(mrun[r] - mnew);
      float p0 = __expf(a0 - mnew);
      float p1 = __expf(a1 - mnew);
      float rs = p0 + p1;
      rs += SWIZ_XOR(rs, 1);
      rs += SWIZ_XOR(rs, 2);
      rs += SWIZ_XOR(rs, 4);
      rs += SWIZ_XOR(rs, 8);
      lrun[r] = lrun[r] * alpha[r] + rs;
      mrun[r] = mnew;
      pl[wv][(r + half * 8) * 32 + l15]      = f2bf_trunc(p0);
      pl[wv][(r + half * 8) * 32 + 16 + l15] = f2bf_trunc(p1);
    }
    __syncthreads();

    #pragma unroll
    for (int ct = 0; ct < 8; ++ct)                   // rescale (alpha VGPR-aligned)
      #pragma unroll
      for (int r = 0; r < 8; ++r) acc[ct][r] *= alpha[r];

    AB pa;                                           // P as A operand, 2x ds_load_b128
    {
      const uint4* pr = (const uint4*)(&pl[wv][l15 * 32]);
      *(uint4*)&pa.u[0] = pr[half ? 2 : 0];
      *(uint4*)&pa.u[4] = pr[half ? 3 : 1];
    }

    #pragma unroll
    for (int ct = 0; ct < 8; ++ct) {                 // O_T += P(16x32) x V^T(32x16)
      AB vt;
      const uint4* vp = (const uint4*)(vs + (ct * 16 + l15) * 16);
      *(uint4*)&vt.u[0] = vp[half ? 2 : 0];
      *(uint4*)&vt.u[4] = vp[half ? 3 : 1];
      acc[ct] = wmma_bf16(pa, vt, acc[ct]);
    }
    __syncthreads();                                 // all waves done with vs/ks
  }

  #pragma unroll
  for (int ct = 0; ct < 8; ++ct) {                   // normalize, add to filtered
    #pragma unroll
    for (int r = 0; r < 8; ++r) {
      int n = r + half * 8;
      int c = (ch0 + ct) * 16 + l15;
      float v = acc[ct][r] / lrun[r];
      float* po = out + ((size_t)b * CC + c) * HW + n0 + n;
      *po += v;
    }
  }
}

// ---------------------------------------------------------------------------
// Host side
// ---------------------------------------------------------------------------
extern "C" void kernel_launch(void* const* d_in, const int* in_sizes, int n_in,
                              void* d_out, int out_size, void* d_ws, size_t ws_size,
                              hipStream_t stream) {
  const float* x     = (const float*)d_in[0];
  const float* w_adj = (const float*)d_in[1];
  const float* b_adj = (const float*)d_in[2];
  const float* w_q   = (const float*)d_in[3];
  const float* b_q   = (const float*)d_in[4];
  const float* w_k   = (const float*)d_in[5];
  const float* b_k   = (const float*)d_in[6];
  const float* w_v   = (const float*)d_in[7];
  const float* b_v   = (const float*)d_in[8];
  const float* w_c1  = (const float*)d_in[9];
  const float* b_c1  = (const float*)d_in[10];
  const float* w_c2  = (const float*)d_in[11];
  const float* b_c2  = (const float*)d_in[12];
  const float* w_fc  = (const float*)d_in[13];
  const float* b_fc  = (const float*)d_in[14];
  float* out = (float*)d_out;

  char* ws = (char*)d_ws;
  size_t off = 0;
  auto alloc = [&](size_t bytes) -> void* {
    void* p = ws + off;
    off += (bytes + 255) & ~(size_t)255;
    return p;
  };
  float*          xa   = (float*)       alloc((size_t)BN*CC*HW*4);       // 16 MB fp32
  unsigned int*   xpk  = (unsigned int*)alloc((size_t)BN*(CC/2)*HW*4);   //  8 MB packed bf16
  unsigned int*   qpk  = (unsigned int*)alloc((size_t)BN*HW*(QKC/2)*4);  //  1 MB [b][n][qk/2]
  unsigned int*   kpk  = (unsigned int*)alloc((size_t)BN*(QKC/2)*HW*4);  //  1 MB [b][qk/2][m]
  unsigned short* vbuf = (unsigned short*)alloc((size_t)BN*CC*HW*2);     //  8 MB [b][c][m]
  unsigned int*   h1pk = (unsigned int*)alloc((size_t)BN*32*HW*4);       //  2 MB [b][oc/2][hw]
  unsigned int*   wqp  = (unsigned int*)alloc((size_t)QKC*(CC/2)*4);
  unsigned int*   wkp  = (unsigned int*)alloc((size_t)QKC*(CC/2)*4);
  unsigned int*   wvp  = (unsigned int*)alloc((size_t)CC*(CC/2)*4);
  unsigned int*   w1p  = (unsigned int*)alloc((size_t)9*64*(CC/2)*4);
  unsigned int*   w2p  = (unsigned int*)alloc((size_t)9*128*32*4);
  float*          pooled = (float*)alloc((size_t)BN*128*4);
  int*            dtp    = (int*)  alloc((size_t)BN*4);
  float*          sob    = (float*)alloc((size_t)BN*HW*4);
  float*          satb   = (float*)alloc((size_t)BN*HW*4);
  float*          fmag   = (float*)alloc((size_t)BN*HW*4);

  // zero accumulators
  k_zero<<<(BN*128 + 255)/256, 256, 0, stream>>>(pooled, BN*128);
  k_zero<<<(BN*HW  + 255)/256, 256, 0, stream>>>(fmag,   BN*HW);

  // one-time weight packing (bf16 pair dwords, operand-ready)
  k_pack_w<<<(QKC*CC/2 + 255)/256, 256, 0, stream>>>(w_q, wqp, QKC*CC/2);
  k_pack_w<<<(QKC*CC/2 + 255)/256, 256, 0, stream>>>(w_k, wkp, QKC*CC/2);
  k_pack_w<<<(CC*CC/2  + 255)/256, 256, 0, stream>>>(w_v, wvp, CC*CC/2);
  k_pack_w3<<<(9*64*CC/2 + 255)/256, 256, 0, stream>>>(w_c1, w1p, 64, CC);
  k_pack_w3<<<(9*128*32  + 255)/256, 256, 0, stream>>>(w_c2, w2p, 128, 64);

  // xa (fp32 + packed bf16)
  k_adjust<<<(BN*(CC/2)*HW)/256, 256, 0, stream>>>(x, w_adj, b_adj, xa, xpk);

  // q/k/v 1x1 GEMMs (WMMA, register-blocked)
  k_gemm<2,1><<<dim3(HW/16, 1, BN), 32, 0, stream>>>(wqp, b_q, xpk, qpk,  QKC);
  k_gemm<2,2><<<dim3(HW/16, 1, BN), 32, 0, stream>>>(wkp, b_k, xpk, kpk,  QKC);
  k_gemm<4,0><<<dim3(HW/16, 4, BN), 32, 0, stream>>>(wvp, b_v, xpk, vbuf, CC);

  // classifier conv3x3 implicit GEMMs (WMMA, register-blocked over all oc)
  k_conv3x3<4><<<dim3(HW/16, 1, BN), 32, 0, stream>>>(w1p, b_c1, xpk,  h1pk, nullptr, CC, 64, 0);
  k_conv3x3<8><<<dim3(HW/16, 1, BN), 32, 0, stream>>>(w2p, b_c2, h1pk, nullptr, pooled, 64, 128, 1);

  // logits (-> d_out tail) + routing argmax
  k_logits<<<1, 256, 0, stream>>>(pooled, w_fc, b_fc, out + (size_t)BN*CC*HW, dtp);

  // experts
  k_experts<<<(BN*HW)/256, 256, 0, stream>>>(xa, sob, satb);
  k_fft<<<BN*CC, 64, 0, stream>>>(xa, fmag);

  // routed dispatch into d_out
  k_filtered<<<(BN*CC*HW)/256, 256, 0, stream>>>(xa, sob, satb, fmag, dtp, out);

  // flash attention accumulates on top of filtered
  // grid: (n-supertiles, channel half, batch), 4 waves per block
  k_attn<<<dim3(HW/(16*ATT_WAVES), 2, BN), 32*ATT_WAVES, 0, stream>>>(qpk, kpk, vbuf, out);
}
